// GAT_67388036874853
// MI455X (gfx1250) — matrix operand order repaired and checked by reading, not compiled, not run on hardware
//
#include <hip/hip_runtime.h>
#include <hip/hip_bf16.h>
#include <math.h>

// ---------------------------------------------------------------------------
// GAT (3-layer) for MI455X / gfx1250.
//   - Dense linear layers via V_WMMA_F32_16X16X32_F16 (wave32 WMMA),
//     fragments fed from LDS with ds_load_b128 (vectorized, bank-conflict-free).
//   - Edge softmax via order-preserving-uint atomicMax + atomicAdd.
//   - Attention-weighted scatter-sum: one wave per edge, f32 atomics.
// ---------------------------------------------------------------------------

typedef __attribute__((ext_vector_type(16))) _Float16 v16h;
typedef __attribute__((ext_vector_type(8)))  _Float16 v8h;
typedef __attribute__((ext_vector_type(8)))  float    v8f;

#define KDIM 128            // inner dim of every GEMM in this net
#define TILE_ROWS 64        // rows per block (2 waves x 32 rows)
#define LDS_PAD 8           // halfs: keeps 16B alignment, kills bank conflicts
#define AROW (KDIM + LDS_PAD)
#define NEG_SLOPE 0.2f

// ---------------------------------------------------------------------------
// WMMA GEMM: Out[n, NCOLS] = A[n, 128] * W[128, NCOLS]   (f32 in, f16 MMA,
// f32 accumulate).  Block = 64 threads = 2 waves; each wave owns 32 rows
// (two 16-row C tiles sharing each B fragment).
// ---------------------------------------------------------------------------
template<int NCOLS>
__global__ __launch_bounds__(64)
void gemm_wmma_f16(const float* __restrict__ A, const float* __restrict__ W,
                   float* __restrict__ Out, int nrows)
{
    __shared__ _Float16 aT[TILE_ROWS][AROW];   // 64 x 136 halfs = 17 KB
    __shared__ _Float16 wT[NCOLS][AROW];       // W transposed, <= 34 KB

    const int tid     = threadIdx.x;
    const int rowBase = blockIdx.x * TILE_ROWS;

    // Stage W transposed (f32 -> f16): wT[col][k] = W[k*NCOLS + col].
    for (int i = tid; i < KDIM * NCOLS; i += 64) {
        int k = i / NCOLS, n = i % NCOLS;
        wT[n][k] = (_Float16)W[i];
    }
    // Stage activation tile via float4 global loads (coalesced b128).
    for (int i = tid; i < TILE_ROWS * (KDIM / 4); i += 64) {
        int r  = i / (KDIM / 4);
        int k4 = (i % (KDIM / 4)) * 4;
        int gr = rowBase + r;
        float4 v = make_float4(0.f, 0.f, 0.f, 0.f);
        if (gr < nrows)
            v = reinterpret_cast<const float4*>(A + (size_t)gr * KDIM)[k4 >> 2];
        aT[r][k4 + 0] = (_Float16)v.x;
        aT[r][k4 + 1] = (_Float16)v.y;
        aT[r][k4 + 2] = (_Float16)v.z;
        aT[r][k4 + 3] = (_Float16)v.w;
    }
    __syncthreads();

    const int wave = tid >> 5;
    const int lane = tid & 31;
    const int m    = lane & 15;      // row/col within the 16x16 tile
    const int hi   = lane >> 4;      // lane-half selector
    const int r0   = wave * 32;      // this wave's first row

    constexpr int NT = NCOLS / 16;
    v8f acc0[NT], acc1[NT];
    #pragma unroll
    for (int t = 0; t < NT; ++t)
        #pragma unroll
        for (int i = 0; i < 8; ++i) { acc0[t][i] = 0.f; acc1[t][i] = 0.f; }

    #pragma unroll
    for (int kc = 0; kc < KDIM; kc += 32) {
        // A fragment (16x32 f16): lane holds K = kc+hi*8..+7 and kc+16+hi*8..+7
        // of its row -> two contiguous, 16B-aligned ds_load_b128 each.
        v8h a0lo = *(const v8h*)&aT[r0 + m][kc + hi * 8];
        v8h a0hi = *(const v8h*)&aT[r0 + m][kc + 16 + hi * 8];
        v8h a1lo = *(const v8h*)&aT[r0 + 16 + m][kc + hi * 8];
        v8h a1hi = *(const v8h*)&aT[r0 + 16 + m][kc + 16 + hi * 8];
        v16h af0 = __builtin_shufflevector(a0lo, a0hi,
                    0,1,2,3,4,5,6,7,8,9,10,11,12,13,14,15);
        v16h af1 = __builtin_shufflevector(a1lo, a1hi,
                    0,1,2,3,4,5,6,7,8,9,10,11,12,13,14,15);
        #pragma unroll
        for (int t = 0; t < NT; ++t) {
            // B fragment (32x16 f16): lane holds K = kc+hi*8..+7 and kc+16+hi*8..+7
            // of column t*16+m -> contiguous in transposed wT, two ds_load_b128.
            v8h blo = *(const v8h*)&wT[t * 16 + m][kc + hi * 8];
            v8h bhi = *(const v8h*)&wT[t * 16 + m][kc + 16 + hi * 8];
            v16h bf = __builtin_shufflevector(blo, bhi,
                        0,1,2,3,4,5,6,7,8,9,10,11,12,13,14,15);
            acc0[t] = __builtin_amdgcn_wmma_f32_16x16x32_f16(
                false, af0, false, bf, (short)0, acc0[t], false, false);
            acc1[t] = __builtin_amdgcn_wmma_f32_16x16x32_f16(
                false, af1, false, bf, (short)0, acc1[t], false, false);
        }
    }

    // C layout: VGPR r holds M=r (lanes 0-15) / M=r+8 (lanes 16-31), N=m.
    #pragma unroll
    for (int t = 0; t < NT; ++t)
        #pragma unroll
        for (int r = 0; r < 8; ++r) {
            int g0 = rowBase + r0 + r + hi * 8;
            int g1 = g0 + 16;
            if (g0 < nrows) Out[(size_t)g0 * NCOLS + t * 16 + m] = acc0[t][r];
            if (g1 < nrows) Out[(size_t)g1 * NCOLS + t * 16 + m] = acc1[t][r];
        }
}

// ---------------------------------------------------------------------------
// Per-node attention logits: als[n,h] = <h[n,h,:], a_s[h,:]>, same for ald.
// ---------------------------------------------------------------------------
__global__ void logits_kernel(const float* __restrict__ h,
                              const float* __restrict__ a_s,
                              const float* __restrict__ a_d,
                              float* __restrict__ als, float* __restrict__ ald,
                              int n, int heads, int c)
{
    int idx = blockIdx.x * blockDim.x + threadIdx.x;
    if (idx >= n * heads) return;
    int node = idx / heads, hd = idx % heads;
    const float* hp = h + (size_t)node * heads * c + hd * c;
    const float* as = a_s + hd * c;
    const float* ad = a_d + hd * c;
    float ss = 0.f, sd = 0.f;
    for (int i = 0; i < c; ++i) { float v = hp[i]; ss += v * as[i]; sd += v * ad[i]; }
    als[idx] = ss;
    ald[idx] = sd;
}

// Order-preserving key so atomicMax(uint) == max(float).
__device__ __forceinline__ unsigned fkey(float f) {
    unsigned u = __float_as_uint(f);
    return (u & 0x80000000u) ? ~u : (u | 0x80000000u);
}
__device__ __forceinline__ float funkey(unsigned k) {
    unsigned u = (k & 0x80000000u) ? (k & 0x7FFFFFFFu) : ~k;
    return __uint_as_float(u);
}

__device__ __forceinline__ void edge_nodes(const int* ei, int e, int E_orig,
                                           int& s, int& d) {
    if (e < E_orig) { s = ei[e]; d = ei[E_orig + e]; }
    else            { s = d = e - E_orig; }          // self-loop
}

__global__ void fill_f32(float* p, float v, size_t n) {
    size_t i = (size_t)blockIdx.x * blockDim.x + threadIdx.x;
    if (i < n) p[i] = v;
}
__global__ void fill_u32(unsigned* p, unsigned v, size_t n) {
    size_t i = (size_t)blockIdx.x * blockDim.x + threadIdx.x;
    if (i < n) p[i] = v;
}

// Pass 1: segment max of leaky-relu logits over dst.
__global__ void edge_max_kernel(const int* __restrict__ ei, int E_total, int E_orig,
                                int heads, const float* __restrict__ als,
                                const float* __restrict__ ald,
                                unsigned* __restrict__ mkey)
{
    int idx = blockIdx.x * blockDim.x + threadIdx.x;
    if (idx >= E_total * heads) return;
    int e = idx / heads, hd = idx % heads;
    int s, d; edge_nodes(ei, e, E_orig, s, d);
    float v = als[s * heads + hd] + ald[d * heads + hd];
    v = (v > 0.f) ? v : NEG_SLOPE * v;
    atomicMax(&mkey[d * heads + hd], fkey(v));
}

// Pass 2: segment sum of exp(e - max).
__global__ void edge_sum_kernel(const int* __restrict__ ei, int E_total, int E_orig,
                                int heads, const float* __restrict__ als,
                                const float* __restrict__ ald,
                                const unsigned* __restrict__ mkey,
                                float* __restrict__ z)
{
    int idx = blockIdx.x * blockDim.x + threadIdx.x;
    if (idx >= E_total * heads) return;
    int e = idx / heads, hd = idx % heads;
    int s, d; edge_nodes(ei, e, E_orig, s, d);
    float v = als[s * heads + hd] + ald[d * heads + hd];
    v = (v > 0.f) ? v : NEG_SLOPE * v;
    float mm = funkey(mkey[d * heads + hd]);
    atomicAdd(&z[d * heads + hd], expf(v - mm));
}

// Pass 3: accum[dst] += alpha * h[src].  One wave per edge (coalesced).
__global__ void edge_agg_kernel(const int* __restrict__ ei, int E_total, int E_orig,
                                int heads, int c,
                                const float* __restrict__ h,
                                const float* __restrict__ als,
                                const float* __restrict__ ald,
                                const unsigned* __restrict__ mkey,
                                const float* __restrict__ z,
                                float* __restrict__ accum)
{
    int wv   = (blockIdx.x * blockDim.x + threadIdx.x) >> 5;
    int lane = threadIdx.x & 31;
    if (wv >= E_total) return;
    int s, d; edge_nodes(ei, wv, E_orig, s, d);
    int F = heads * c;
    for (int ch = lane; ch < F; ch += 32) {
        int hd = ch / c;
        float v = als[s * heads + hd] + ald[d * heads + hd];
        v = (v > 0.f) ? v : NEG_SLOPE * v;
        float mm    = funkey(mkey[d * heads + hd]);
        float alpha = expf(v - mm) / (z[d * heads + hd] + 1e-16f);
        atomicAdd(&accum[(size_t)d * F + ch], h[(size_t)s * F + ch] * alpha);
    }
}

// Bias + optional ELU, in place.
__global__ void finalize_kernel(float* __restrict__ buf, const float* __restrict__ bias,
                                int n, int F, int do_elu)
{
    size_t idx = (size_t)blockIdx.x * blockDim.x + threadIdx.x;
    if (idx >= (size_t)n * F) return;
    float v = buf[idx] + bias[idx % F];
    if (do_elu) v = (v > 0.f) ? v : (expf(v) - 1.0f);
    buf[idx] = v;
}

// ---------------------------------------------------------------------------
extern "C" void kernel_launch(void* const* d_in, const int* in_sizes, int n_in,
                              void* d_out, int out_size, void* d_ws, size_t ws_size,
                              hipStream_t stream)
{
    const float* x   = (const float*)d_in[0];
    const int*   ei  = (const int*)  d_in[1];
    const float* W1  = (const float*)d_in[4];
    const float* a1s = (const float*)d_in[5];
    const float* a1d = (const float*)d_in[6];
    const float* b1  = (const float*)d_in[7];
    const float* W2  = (const float*)d_in[8];
    const float* a2s = (const float*)d_in[9];
    const float* a2d = (const float*)d_in[10];
    const float* b2  = (const float*)d_in[11];
    const float* W3  = (const float*)d_in[12];
    const float* a3s = (const float*)d_in[13];
    const float* a3d = (const float*)d_in[14];
    const float* b3  = (const float*)d_in[15];
    float* out = (float*)d_out;

    const int Nn = in_sizes[0] / KDIM;   // 50000
    const int Eo = in_sizes[1] / 2;      // 800000
    const int Et = Eo + Nn;              // + self loops
    const int Hh = 4, Cc = 32, F = Hh * Cc;

    // Workspace carve (all 4-byte types).
    float*    bufH   = (float*)d_ws;                  // [Nn,128] GEMM out
    float*    bufAgg = bufH   + (size_t)Nn * F;       // [Nn,128] aggregation
    float*    als    = bufAgg + (size_t)Nn * F;       // [Nn,H]
    float*    ald    = als    + (size_t)Nn * Hh;      // [Nn,H]
    unsigned* mkey   = (unsigned*)(ald + (size_t)Nn * Hh);
    float*    zbuf   = (float*)(mkey + (size_t)Nn * Hh);

    const int gemmGrid  = (Nn + TILE_ROWS - 1) / TILE_ROWS;
    const int nodeHGrid = (Nn * Hh + 255) / 256;
    const int edgeHGrid = (Et * Hh + 255) / 256;
    const int edgeWGrid = (Et * 32 + 127) / 128;      // wave per edge
    const int bigFill   = (int)(((size_t)Nn * F + 255) / 256);
    const int smallFill = (Nn * Hh + 255) / 256;

    // ---------------- Layer 1 (128 -> 4x32, concat, ELU) ----------------
    gemm_wmma_f16<128><<<gemmGrid, 64, 0, stream>>>(x, W1, bufH, Nn);
    logits_kernel<<<nodeHGrid, 256, 0, stream>>>(bufH, a1s, a1d, als, ald, Nn, Hh, Cc);
    fill_f32<<<bigFill,   256, 0, stream>>>(bufAgg, 0.f, (size_t)Nn * F);
    fill_u32<<<smallFill, 256, 0, stream>>>(mkey, 0u, (size_t)Nn * Hh);
    fill_f32<<<smallFill, 256, 0, stream>>>(zbuf, 0.f, (size_t)Nn * Hh);
    edge_max_kernel<<<edgeHGrid, 256, 0, stream>>>(ei, Et, Eo, Hh, als, ald, mkey);
    edge_sum_kernel<<<edgeHGrid, 256, 0, stream>>>(ei, Et, Eo, Hh, als, ald, mkey, zbuf);
    edge_agg_kernel<<<edgeWGrid, 128, 0, stream>>>(ei, Et, Eo, Hh, Cc, bufH, als, ald,
                                                   mkey, zbuf, bufAgg);
    finalize_kernel<<<bigFill, 256, 0, stream>>>(bufAgg, b1, Nn, F, 1);

    // ---------------- Layer 2 (128 -> 4x32, concat, ELU) ----------------
    gemm_wmma_f16<128><<<gemmGrid, 64, 0, stream>>>(bufAgg, W2, bufH, Nn);
    logits_kernel<<<nodeHGrid, 256, 0, stream>>>(bufH, a2s, a2d, als, ald, Nn, Hh, Cc);
    fill_f32<<<bigFill,   256, 0, stream>>>(bufAgg, 0.f, (size_t)Nn * F);
    fill_u32<<<smallFill, 256, 0, stream>>>(mkey, 0u, (size_t)Nn * Hh);
    fill_f32<<<smallFill, 256, 0, stream>>>(zbuf, 0.f, (size_t)Nn * Hh);
    edge_max_kernel<<<edgeHGrid, 256, 0, stream>>>(ei, Et, Eo, Hh, als, ald, mkey);
    edge_sum_kernel<<<edgeHGrid, 256, 0, stream>>>(ei, Et, Eo, Hh, als, ald, mkey, zbuf);
    edge_agg_kernel<<<edgeWGrid, 128, 0, stream>>>(ei, Et, Eo, Hh, Cc, bufH, als, ald,
                                                   mkey, zbuf, bufAgg);
    finalize_kernel<<<bigFill, 256, 0, stream>>>(bufAgg, b2, Nn, F, 1);

    // ---------------- Layer 3 (128 -> 32, heads=1, no ELU) ---------------
    const int edge1Grid = (Et + 255) / 256;
    const int node1Grid = (Nn + 255) / 256;
    const int outFill   = (int)(((size_t)Nn * Cc + 255) / 256);
    gemm_wmma_f16<32><<<gemmGrid, 64, 0, stream>>>(bufAgg, W3, bufH, Nn);
    logits_kernel<<<node1Grid, 256, 0, stream>>>(bufH, a3s, a3d, als, ald, Nn, 1, Cc);
    fill_f32<<<outFill,   256, 0, stream>>>(out, 0.f, (size_t)Nn * Cc);
    fill_u32<<<node1Grid, 256, 0, stream>>>(mkey, 0u, (size_t)Nn);
    fill_f32<<<node1Grid, 256, 0, stream>>>(zbuf, 0.f, (size_t)Nn);
    edge_max_kernel<<<edge1Grid, 256, 0, stream>>>(ei, Et, Eo, 1, als, ald, mkey);
    edge_sum_kernel<<<edge1Grid, 256, 0, stream>>>(ei, Et, Eo, 1, als, ald, mkey, zbuf);
    edge_agg_kernel<<<edgeWGrid, 128, 0, stream>>>(ei, Et, Eo, 1, Cc, bufH, als, ald,
                                                   mkey, zbuf, out);
    finalize_kernel<<<outFill, 256, 0, stream>>>(out, b3, Nn, Cc, 0);
}
